// DispersiveLossV2_52948356825714
// MI455X (gfx1250) — matrix operand order repaired and checked by reading, not compiled, not run on hardware
//
#include <hip/hip_runtime.h>
#include <hip/hip_bf16.h>

typedef __attribute__((ext_vector_type(16))) __bf16 v16bf;
typedef __attribute__((ext_vector_type(4)))  __bf16 v4bf;
typedef __attribute__((ext_vector_type(8)))  float  v8f;

#define NROWS 1024
#define LAMBDA_DISP 0.25f
#define LDSS 40   // LDS row stride in elements (padded, 80B = 16B aligned)

#if defined(__HIP_DEVICE_COMPILE__) && defined(__gfx1250__)
#define DL_GFX1250 1
#else
#define DL_GFX1250 0
#endif

// ---------------------------------------------------------------------------
// async global -> LDS copy of 16 bytes per lane (CDNA5 ASYNCcnt path)
// ---------------------------------------------------------------------------
__device__ __forceinline__ void dl_async16(const __bf16* g, __bf16* l) {
#if DL_GFX1250
    unsigned lds32 = (unsigned)(unsigned long long)l;  // addr[31:0] == LDS offset
    unsigned long long ga = (unsigned long long)g;
    asm volatile("global_load_async_to_lds_b128 %0, %1, off"
                 :: "v"(lds32), "v"(ga) : "memory");
#else
    *(uint4*)l = *(const uint4*)g;
#endif
}

__device__ __forceinline__ void dl_wait_async() {
#if DL_GFX1250
    asm volatile("s_wait_asynccnt 0x0" ::: "memory");
#endif
}

// ---------------------------------------------------------------------------
// Kernel 1: per-row sum of squares -> inv_norm, sq
// ---------------------------------------------------------------------------
__global__ __launch_bounds__(256)
void dl_rownorm(const float* __restrict__ z, float* __restrict__ inv,
                float* __restrict__ sq, int K) {
    __shared__ float red[256];
    const int row = blockIdx.x;
    const float4* z4 = (const float4*)(z + (size_t)row * K);
    const int n4 = K >> 2;
    float s = 0.f;
    for (int k = threadIdx.x; k < n4; k += 256) {
        float4 v = z4[k];
        s += v.x * v.x + v.y * v.y + v.z * v.z + v.w * v.w;
    }
    red[threadIdx.x] = s;
    __syncthreads();
    for (int off = 128; off > 0; off >>= 1) {
        if (threadIdx.x < off) red[threadIdx.x] += red[threadIdx.x + off];
        __syncthreads();
    }
    if (threadIdx.x == 0) {
        float ss  = red[0];
        float iv  = 1.f / fmaxf(sqrtf(ss), 1e-12f);
        inv[row]  = iv;
        sq[row]   = ss * iv * iv;   // |z_i|^2 after normalization (~1)
    }
}

// ---------------------------------------------------------------------------
// Kernel 2: normalize + split fp32 -> bf16 hi + bf16 lo   (z ~= hi + lo)
// ---------------------------------------------------------------------------
__global__ __launch_bounds__(256)
void dl_convert(const float* __restrict__ z, const float* __restrict__ inv,
                __bf16* __restrict__ hi, __bf16* __restrict__ lo,
                int kshift, int use_lo, size_t n4) {
    size_t i = (size_t)blockIdx.x * 256 + threadIdx.x;   // float4 index
    if (i >= n4) return;
    float4 v = ((const float4*)z)[i];
    size_t e0 = i * 4;
    int row = (int)(e0 >> kshift);
    float iv = inv[row];
    float a0 = v.x * iv, a1 = v.y * iv, a2 = v.z * iv, a3 = v.w * iv;
    __bf16 h0 = (__bf16)a0, h1 = (__bf16)a1, h2 = (__bf16)a2, h3 = (__bf16)a3;
    v4bf h = {h0, h1, h2, h3};
    *(v4bf*)(hi + e0) = h;
    if (use_lo) {
        v4bf l = {(__bf16)(a0 - (float)h0), (__bf16)(a1 - (float)h1),
                  (__bf16)(a2 - (float)h2), (__bf16)(a3 - (float)h3)};
        *(v4bf*)(lo + e0) = l;
    }
}

// ---------------------------------------------------------------------------
// Kernel 3: Gram GEMM  G += Zn . Zn^T  via v_wmma_f32_16x16x32_bf16
//   grid (8,8,16): x = M tile (128 rows), y = N tile, z = K slice (K/16)
//   256 threads = 8 wave32 waves; each wave computes a 32x64 sub-tile.
//   Double-buffered LDS, tiles staged with global_load_async_to_lds_b128.
// ---------------------------------------------------------------------------
__device__ __forceinline__ v16bf dl_frag(const __bf16* p, int o0, int o1) {
    union { uint4 u[2]; v16bf v; } t;
    t.u[0] = *(const uint4*)(p + o0);
    t.u[1] = *(const uint4*)(p + o1);
    return t.v;
}

template <int USE_LO>
__global__ __launch_bounds__(256)
void dl_gram(const __bf16* __restrict__ hi, const __bf16* __restrict__ lo,
             float* __restrict__ G, int K, int kslice) {
    constexpr int TSZ = 128 * LDSS;
    __shared__ __align__(16) __bf16 sAh[2 * TSZ];
    __shared__ __align__(16) __bf16 sBh[2 * TSZ];
    __shared__ __align__(16) __bf16 sAl[USE_LO ? 2 * TSZ : 16];
    __shared__ __align__(16) __bf16 sBl[USE_LO ? 2 * TSZ : 16];

    const int tid  = threadIdx.x;
    const int lane = tid & 31;
    const int w    = tid >> 5;     // wave 0..7
    const int wm   = w & 3;        // 4 row groups of 32
    const int wn   = w >> 2;       // 2 col groups of 64
    const int m0   = blockIdx.x * 128;
    const int n0   = blockIdx.y * 128;
    const int kb0  = blockIdx.z * kslice;

    // cooperative tile loader: thread -> (row, 16-elem half)
    const int ldr  = tid >> 1;
    const int ldc  = (tid & 1) * 16;
    const size_t rowA = (size_t)(m0 + ldr) * K + ldc;   // + kg
    const size_t rowB = (size_t)(n0 + ldr) * K + ldc;
    const int     lof = ldr * LDSS + ldc;               // LDS dest offset

    auto issue = [&](int kg, int buf) {
        const __bf16* gA = hi + rowA + kg;
        const __bf16* gB = hi + rowB + kg;
        __bf16* lA = sAh + buf * TSZ + lof;
        __bf16* lB = sBh + buf * TSZ + lof;
        dl_async16(gA, lA);  dl_async16(gA + 8, lA + 8);
        dl_async16(gB, lB);  dl_async16(gB + 8, lB + 8);
        if constexpr (USE_LO) {
            const __bf16* qA = lo + rowA + kg;
            const __bf16* qB = lo + rowB + kg;
            __bf16* eA = sAl + buf * TSZ + lof;
            __bf16* eB = sBl + buf * TSZ + lof;
            dl_async16(qA, eA);  dl_async16(qA + 8, eA + 8);
            dl_async16(qB, eB);  dl_async16(qB + 8, eB + 8);
        }
    };

    v8f acc[2][4];
#pragma unroll
    for (int mt = 0; mt < 2; ++mt)
#pragma unroll
        for (int nt = 0; nt < 4; ++nt)
#pragma unroll
            for (int r = 0; r < 8; ++r) acc[mt][nt][r] = 0.f;

    // fragment addressing (ISA 7.12.2, wave32 bf16 16x16x32 layouts)
    const int khalf = (lane >> 4) * 8;    // A: chunks at khalf, khalf+16
    const int bkoff = (lane >> 4) * 8;    // B: chunks at bkoff, bkoff+16
    const int frow  = lane & 15;

    const int nk = kslice >> 5;           // K-steps of 32

    issue(kb0, 0);
    dl_wait_async();
    __syncthreads();

    for (int it = 0; it < nk; ++it) {
        const int cur = it & 1;
        if (it + 1 < nk) issue(kb0 + (it + 1) * 32, cur ^ 1);

        const __bf16* bAh = sAh + cur * TSZ;
        const __bf16* bBh = sBh + cur * TSZ;
        const __bf16* bAl = sAl + (USE_LO ? cur * TSZ : 0);
        const __bf16* bBl = sBl + (USE_LO ? cur * TSZ : 0);

        v16bf ah[2], al[2];
#pragma unroll
        for (int mt = 0; mt < 2; ++mt) {
            int r = (wm * 32 + mt * 16 + frow) * LDSS;
            ah[mt] = dl_frag(bAh + r, khalf, khalf + 16);
            if constexpr (USE_LO) al[mt] = dl_frag(bAl + r, khalf, khalf + 16);
        }
#pragma unroll
        for (int nt = 0; nt < 4; ++nt) {
            int c = (wn * 64 + nt * 16 + frow) * LDSS;
            v16bf bh = dl_frag(bBh + c, bkoff, bkoff + 16);
#pragma unroll
            for (int mt = 0; mt < 2; ++mt)
                acc[mt][nt] = __builtin_amdgcn_wmma_f32_16x16x32_bf16(
                    false, ah[mt], false, bh, (short)0, acc[mt][nt], false, false);
            if constexpr (USE_LO) {
                v16bf bl = dl_frag(bBl + c, bkoff, bkoff + 16);
#pragma unroll
                for (int mt = 0; mt < 2; ++mt) {
                    acc[mt][nt] = __builtin_amdgcn_wmma_f32_16x16x32_bf16(
                        false, ah[mt], false, bl, (short)0, acc[mt][nt], false, false);
                    acc[mt][nt] = __builtin_amdgcn_wmma_f32_16x16x32_bf16(
                        false, al[mt], false, bh, (short)0, acc[mt][nt], false, false);
                }
            }
        }

        if (it + 1 < nk) {
            dl_wait_async();
            __syncthreads();
        }
    }

    // C/D layout: lanes 0-15 -> N=lane, M=r; lanes 16-31 -> N=lane-16, M=r+8
    const int mlane = (lane >> 4) << 3;
    const int nlane = lane & 15;
#pragma unroll
    for (int mt = 0; mt < 2; ++mt)
#pragma unroll
        for (int nt = 0; nt < 4; ++nt) {
            int M = m0 + wm * 32 + mt * 16 + mlane;
            int N = n0 + wn * 64 + nt * 16 + nlane;
#pragma unroll
            for (int r = 0; r < 8; ++r)
                atomicAdd(&G[(size_t)(M + r) * NROWS + N], acc[mt][nt][r]);
        }
}

// ---------------------------------------------------------------------------
// Kernel 4: e = exp(-max(sq_i+sq_j-2G,0)/TAU), partial sums per block
// ---------------------------------------------------------------------------
__global__ __launch_bounds__(256)
void dl_reduce(const float* __restrict__ G, const float* __restrict__ sq,
               float* __restrict__ partials) {
    __shared__ float red[256];
    float s = 0.f;
#pragma unroll
    for (int j = 0; j < 4; ++j) {
        size_t idx = (size_t)blockIdx.x * 1024 + (size_t)j * 256 + threadIdx.x;
        int i  = (int)(idx >> 10);
        int jj = (int)(idx & 1023);
        float d2 = fmaxf(sq[i] + sq[jj] - 2.f * G[idx], 0.f);
        s += __expf(-d2);   // TAU == 1
    }
    red[threadIdx.x] = s;
    __syncthreads();
    for (int off = 128; off > 0; off >>= 1) {
        if (threadIdx.x < off) red[threadIdx.x] += red[threadIdx.x + off];
        __syncthreads();
    }
    if (threadIdx.x == 0) partials[blockIdx.x] = red[0];
}

// ---------------------------------------------------------------------------
// Kernel 5: final reduce + loss
// ---------------------------------------------------------------------------
__global__ __launch_bounds__(256)
void dl_final(const float* __restrict__ partials, float* __restrict__ out) {
    __shared__ float red[256];
    float s = 0.f;
    for (int i = threadIdx.x; i < 1024; i += 256) s += partials[i];
    red[threadIdx.x] = s;
    __syncthreads();
    for (int off = 128; off > 0; off >>= 1) {
        if (threadIdx.x < off) red[threadIdx.x] += red[threadIdx.x + off];
        __syncthreads();
    }
    if (threadIdx.x == 0) {
        float Bf = (float)NROWS;
        float mean = (red[0] - Bf) / (Bf * (Bf - 1.f));
        out[0] = LAMBDA_DISP * logf(mean);
    }
}

// ---------------------------------------------------------------------------
extern "C" void kernel_launch(void* const* d_in, const int* in_sizes, int n_in,
                              void* d_out, int out_size, void* d_ws, size_t ws_size,
                              hipStream_t stream) {
    const float* z = (const float*)d_in[0];
    float* out = (float*)d_out;
    const size_t n = (size_t)in_sizes[0];       // 1024 * 64 * 1024
    const int K = (int)(n / NROWS);             // 65536
    int kshift = 0;
    while ((1u << kshift) < (unsigned)K) ++kshift;

    char* ws = (char*)d_ws;
    float* G        = (float*)ws;                                   // 4 MB
    float* inv      = (float*)(ws + (size_t)4 * 1024 * 1024);
    float* sq       = inv + NROWS;
    float* partials = sq + NROWS;
    __bf16* hi = (__bf16*)(ws + (size_t)8 * 1024 * 1024);
    const size_t zbytes = n * sizeof(__bf16);                       // 128 MB
    __bf16* lo = (__bf16*)((char*)hi + zbytes);
    const int use_lo =
        (ws_size >= (size_t)8 * 1024 * 1024 + 2 * zbytes) ? 1 : 0;

    hipMemsetAsync(G, 0, (size_t)NROWS * NROWS * sizeof(float), stream);

    dl_rownorm<<<NROWS, 256, 0, stream>>>(z, inv, sq, K);

    const size_t n4 = n / 4;
    dl_convert<<<(unsigned)((n4 + 255) / 256), 256, 0, stream>>>(
        z, inv, hi, lo, kshift, use_lo, n4);

    dim3 g3(NROWS / 128, NROWS / 128, 16);      // (8,8,16)
    if (use_lo)
        dl_gram<1><<<g3, 256, 0, stream>>>(hi, lo, G, K, K / 16);
    else
        dl_gram<0><<<g3, 256, 0, stream>>>(hi, lo, G, K, K / 16);

    dl_reduce<<<NROWS, 256, 0, stream>>>(G, sq, partials);
    dl_final<<<1, 256, 0, stream>>>(partials, out);
}
